// CrossModalAttention_12816182411348
// MI455X (gfx1250) — matrix operand order repaired and checked by reading, not compiled, not run on hardware
//
#include <hip/hip_runtime.h>
#include <math.h>

// ---------------------------------------------------------------------------
// CrossModalAttention for MI455X (gfx1250), wave32 + WMMA bf16 + TDM.
// B=8 T=4096 DL=DM=1024 H=8 HD=128.  BT = 32768 tokens.
// Six GEMMs (32768x1024 @ 1024x{1024,2048}) with v_wmma_f32_16x16x32_bf16.
//   - gemm_proj: f32 activations, converted to bf16 while staging;
//     register-staged + LDS double-buffered so global loads overlap WMMA.
//   - gemm_out:  bf16 activations streamed by the Tensor Data Mover
//     (tensor_load_to_lds, TENSORcnt ping-pong), f32 weights staged manually.
// Per-token 8x8 head attention is tiny (~2 GFLOP) -> f32 VALU kernel.
// ---------------------------------------------------------------------------

typedef __attribute__((ext_vector_type(16))) __bf16 v16bf;
typedef __attribute__((ext_vector_type(8)))  float  v8f;
typedef __attribute__((ext_vector_type(4)))  unsigned int u32x4;
typedef __attribute__((ext_vector_type(8)))  int          i32x8;
typedef __attribute__((ext_vector_type(4)))  int          i32x4;

#define BT_TOKENS 32768
#define KDIM      1024

#define BM 128
#define BN 128
#define BK 32
#define LDT 17   // u32 per LDS tile row: 16 data (32 bf16) + 1 pad

__device__ __forceinline__ unsigned short f2bf(float x) {
  unsigned int u = __builtin_bit_cast(unsigned int, x);
  unsigned int r = u + 0x7FFFu + ((u >> 16) & 1u);   // round-to-nearest-even
  return (unsigned short)(r >> 16);
}
__device__ __forceinline__ float bf2f(unsigned short h) {
  unsigned int u = ((unsigned int)h) << 16;
  return __builtin_bit_cast(float, u);
}

// ---- shared WMMA tile compute: 8x v_wmma_f32_16x16x32_bf16 per wave --------
__device__ __forceinline__ void wmma_tile(const unsigned int* __restrict__ As,
                                          const unsigned int* __restrict__ Bs,
                                          v8f (&acc)[4][2],
                                          int wrow, int wcol, int lmod, int lhalf)
{
  union Frag { v16bf v; unsigned int u[8]; };
  Frag af[4], bf_[2];
#pragma unroll
  for (int fm = 0; fm < 4; ++fm) {
    int mloc = wrow * 64 + fm * 16 + lmod;
#pragma unroll
    for (int j = 0; j < 8; ++j) {
      // A 16x32 bf16: lanes 0-15 K={0..7,16..23}, lanes 16-31 +8
      int k = ((j >> 2) << 4) + ((j & 3) << 1) + (lhalf << 3);
      af[fm].u[j] = As[mloc * LDT + (k >> 1)];
    }
  }
#pragma unroll
  for (int fn = 0; fn < 2; ++fn) {
    int nloc = wcol * 32 + fn * 16 + lmod;
#pragma unroll
    for (int j = 0; j < 8; ++j) {
      // B 32x16 bf16: lanes 0-15 K={0..7,16..23}, lanes 16-31 +8
      int k = ((j >> 2) << 4) + ((j & 3) << 1) + (lhalf << 3);
      bf_[fn].u[j] = Bs[nloc * LDT + (k >> 1)];
    }
  }
#pragma unroll
  for (int fm = 0; fm < 4; ++fm)
#pragma unroll
    for (int fn = 0; fn < 2; ++fn)
      acc[fm][fn] = __builtin_amdgcn_wmma_f32_16x16x32_bf16(
          false, af[fm].v, false, bf_[fn].v, (short)0, acc[fm][fn], false, false);
}

// ---------------------------------------------------------------------------
// gemm_proj: out_bf16[M,N] = f32 A[M,K] * f32 W[K,N] + bias
// Register-staged loads + double-buffered LDS: loads for tile t+1 are issued
// before the WMMAs of tile t, stores to the other LDS buffer come after.
// ---------------------------------------------------------------------------
__global__ __launch_bounds__(256) void gemm_proj(
    const float* __restrict__ A, const float* __restrict__ W,
    const float* __restrict__ bias, unsigned short* __restrict__ outB,
    int M, int N, int K)
{
  __shared__ unsigned int As[2][BM * LDT];
  __shared__ unsigned int Bs[2][BN * LDT];

  const int row0 = blockIdx.x * BM;
  const int col0 = blockIdx.y * BN;
  const int tid  = threadIdx.x;
  const int wave = tid >> 5, lane = tid & 31;
  const int wrow = wave >> 2, wcol = wave & 3;
  const int lmod = lane & 15, lhalf = lane >> 4;

  v8f acc[4][2];
#pragma unroll
  for (int i = 0; i < 4; ++i)
#pragma unroll
    for (int j = 0; j < 2; ++j)
      acc[i][j] = (v8f){0.f, 0.f, 0.f, 0.f, 0.f, 0.f, 0.f, 0.f};

  float2 aReg[8];
  float  bReg[16];

  auto loadA = [&](int kt) {
#pragma unroll
    for (int r = 0; r < 8; ++r) {
      int i = tid + 256 * r, m = i >> 4, kk2 = i & 15;
      aReg[r] = *(const float2*)(A + (size_t)(row0 + m) * K + kt + kk2 * 2);
    }
  };
  auto loadB = [&](int kt) {
#pragma unroll
    for (int r = 0; r < 16; ++r) {
      int i = tid + 256 * r, k = i >> 7, n = i & 127;
      bReg[r] = W[(size_t)(kt + k) * N + col0 + n];
    }
  };
  auto storeA = [&](int buf) {
#pragma unroll
    for (int r = 0; r < 8; ++r) {
      int i = tid + 256 * r, m = i >> 4, kk2 = i & 15;
      As[buf][m * LDT + kk2] =
          (unsigned int)f2bf(aReg[r].x) | ((unsigned int)f2bf(aReg[r].y) << 16);
    }
  };
  auto storeB = [&](int buf) {
    unsigned short* p = (unsigned short*)Bs[buf];
#pragma unroll
    for (int r = 0; r < 16; ++r) {
      int i = tid + 256 * r, k = i >> 7, n = i & 127;
      p[n * (2 * LDT) + k] = f2bf(bReg[r]);
    }
  };

  loadA(0); loadB(0); storeA(0); storeB(0);
  const int nkt = K / BK;
  for (int t = 0; t < nkt; ++t) {
    __syncthreads();
    const int cur = t & 1;
    if (t + 1 < nkt) {
      loadA((t + 1) * BK);          // global loads in flight during WMMA
      loadB((t + 1) * BK);
      __builtin_prefetch(A + (size_t)(row0 + (tid >> 1)) * K + (t + 2) * BK, 0, 1);
    }
    wmma_tile(As[cur], Bs[cur], acc, wrow, wcol, lmod, lhalf);
    if (t + 1 < nkt) { storeA(cur ^ 1); storeB(cur ^ 1); }
  }

#pragma unroll
  for (int fm = 0; fm < 4; ++fm)
#pragma unroll
    for (int fn = 0; fn < 2; ++fn) {
      int n = col0 + wcol * 32 + fn * 16 + lmod;
      float bv = bias[n];
#pragma unroll
      for (int r = 0; r < 8; ++r) {
        int m = row0 + wrow * 64 + fm * 16 + lhalf * 8 + r;
        outB[(size_t)m * N + n] = f2bf(acc[fm][fn][r] + bv);
      }
    }
}

// ---------------------------------------------------------------------------
// gemm_out: outF[M,N] = bf16 A[M,K] * f32 W[K,N] + bias + resid
// A tiles (128x32 bf16, row-major, stride K) streamed by the Tensor Data
// Mover straight into padded LDS rows (pad_interval=16 DWORDs, pad=1 DWORD
// -> 17-u32 row stride), double-buffered on TENSORcnt.
// ---------------------------------------------------------------------------
__global__ __launch_bounds__(256) void gemm_out(
    const unsigned short* __restrict__ A, const float* __restrict__ W,
    const float* __restrict__ bias, const float* __restrict__ resid,
    float* __restrict__ outF, int M, int N, int K)
{
  __shared__ unsigned int As[2][BM * LDT];
  __shared__ unsigned int Bs[2][BN * LDT];

  const int row0 = blockIdx.x * BM;
  const int col0 = blockIdx.y * BN;
  const int tid  = threadIdx.x;
  const int wave = tid >> 5, lane = tid & 31;
  const int wrow = wave >> 2, wcol = wave & 3;
  const int lmod = lane & 15, lhalf = lane >> 4;

  v8f acc[4][2];
#pragma unroll
  for (int i = 0; i < 4; ++i)
#pragma unroll
    for (int j = 0; j < 2; ++j)
      acc[i][j] = (v8f){0.f, 0.f, 0.f, 0.f, 0.f, 0.f, 0.f, 0.f};

  // ---- TDM issue: one wave DMAs the 128x32 bf16 A tile into LDS ----
  auto issueTdm = [&](int kt, int buf) {
    if (wave == 0) {
      unsigned long long ga =
          (unsigned long long)(size_t)(A + (size_t)row0 * K + kt);
      unsigned int lds = (unsigned int)(size_t)(&As[buf][0]);
      u32x4 g0;
      g0.x = 1u;                                          // count=1 (valid)
      g0.y = lds;                                         // lds_addr
      g0.z = (unsigned int)(ga & 0xFFFFFFFFu);            // global_addr[31:0]
      g0.w = (unsigned int)((ga >> 32) & 0x01FFFFFFu)     // global_addr[56:32]
             | 0x80000000u;                               // type=2 (image)
      i32x8 g1;
      g1[0] = (1 << 16)        // data_size = 2 bytes
            | (1 << 20)        // pad_enable
            | (3 << 22);       // pad_interval = 16 DWORDs (one 32-elem row)
                               // pad_amount = 1 DWORD (bits 31:25 = 0)
      g1[1] = (int)(((unsigned int)K & 0xFFFFu) << 16);   // tensor_dim0 lo16
      g1[2] = (int)((((unsigned int)K >> 16) & 0xFFFFu)   // tensor_dim0 hi16
            | (((unsigned int)M & 0xFFFFu) << 16));       // tensor_dim1 lo16
      g1[3] = (int)((((unsigned int)M >> 16) & 0xFFFFu)   // tensor_dim1 hi16
            | (32u << 16));                               // tile_dim0 = 32 (K)
      g1[4] = 128;                                        // tile_dim1 = 128 rows
      g1[5] = K;                                          // tensor_dim0_stride
      g1[6] = 0;
      g1[7] = 0;
      i32x4 gz4 = {0, 0, 0, 0};
      i32x8 gz8 = {0, 0, 0, 0, 0, 0, 0, 0};
      // clang-23 / amdgpu-toolchain 6-arg form:
      // (g0, g1, g2, g3, g4, cpol)
      __builtin_amdgcn_tensor_load_to_lds(g0, g1, gz4, gz4, gz8, 0);
    }
  };

  float bReg[16];
  auto loadB = [&](int kt) {
#pragma unroll
    for (int r = 0; r < 16; ++r) {
      int i = tid + 256 * r, k = i >> 7, n = i & 127;
      bReg[r] = W[(size_t)(kt + k) * N + col0 + n];
    }
  };
  auto storeB = [&](int buf) {
    unsigned short* p = (unsigned short*)Bs[buf];
#pragma unroll
    for (int r = 0; r < 16; ++r) {
      int i = tid + 256 * r, k = i >> 7, n = i & 127;
      p[n * (2 * LDT) + k] = f2bf(bReg[r]);
    }
  };

  issueTdm(0, 0);
  loadB(0); storeB(0);
  const int nkt = K / BK;
  for (int t = 0; t < nkt; ++t) {
    const int cur = t & 1;
    if (t + 1 < nkt) {
      issueTdm((t + 1) * BK, cur ^ 1);   // DMA next A tile into other buffer
      loadB((t + 1) * BK);               // next W tile into registers
    }
    if (wave == 0) {
      if (t + 1 < nkt) __builtin_amdgcn_s_wait_tensorcnt(1);  // tile t done
      else             __builtin_amdgcn_s_wait_tensorcnt(0);
    }
    __syncthreads();                     // A tile in LDS, B stores visible
    wmma_tile(As[cur], Bs[cur], acc, wrow, wcol, lmod, lhalf);
    if (t + 1 < nkt) storeB(cur ^ 1);
    __syncthreads();                     // reads done before buf reuse
  }

#pragma unroll
  for (int fm = 0; fm < 4; ++fm)
#pragma unroll
    for (int fn = 0; fn < 2; ++fn) {
      int n = col0 + wcol * 32 + fn * 16 + lmod;
      float bv = bias[n];
#pragma unroll
      for (int r = 0; r < 8; ++r) {
        int m = row0 + wrow * 64 + fm * 16 + lhalf * 8 + r;
        size_t idx = (size_t)m * N + n;
        outF[idx] = acc[fm][fn][r] + bv + resid[idx];
      }
    }
}

// ---------------------------------------------------------------------------
// Per-token 8x8 head attention (scores over the head axis).
//  Q: (BT, 8, 128) bf16   KV: (BT, 8, 256) bf16 (k then v)   O: (BT,8,128)
// ---------------------------------------------------------------------------
__global__ __launch_bounds__(128) void head_attn(
    const unsigned short* __restrict__ Q,
    const unsigned short* __restrict__ KV,
    unsigned short* __restrict__ O)
{
  const int t = blockIdx.x;
  const int tid = threadIdx.x;

  __shared__ float q[8][128], kk[8][128], vv[8][128];
  __shared__ float sc[8][8];
  __shared__ float wgt[8][9];

  const unsigned short* qp  = Q  + (size_t)t * 1024;
  const unsigned short* kvp = KV + (size_t)t * 2048;

#pragma unroll
  for (int r = 0; r < 8; ++r) {
    int i = tid + 128 * r;
    q[i >> 7][i & 127] = bf2f(qp[i]);
  }
#pragma unroll
  for (int r = 0; r < 16; ++r) {
    int i = tid + 128 * r;
    int h = i >> 8, d2 = i & 255;
    float f = bf2f(kvp[i]);
    if (d2 < 128) kk[h][d2] = f;
    else          vv[h][d2 - 128] = f;
  }
  __syncthreads();

  if (tid < 64) {
    int h = tid >> 3, g = tid & 7;
    float s = 0.f;
#pragma unroll 8
    for (int d = 0; d < 128; ++d) s += q[h][d] * kk[g][d];
    sc[h][g] = s * 0.08838834764831845f;   // 1/sqrt(128)
  }
  __syncthreads();

  if (tid < 8) {
    int h = tid;
    float mx = sc[h][0];
#pragma unroll
    for (int g = 1; g < 8; ++g) mx = fmaxf(mx, sc[h][g]);
    float e[8], sum = 0.f;
#pragma unroll
    for (int g = 0; g < 8; ++g) { e[g] = __expf(sc[h][g] - mx); sum += e[g]; }
    float inv = 1.f / sum;
#pragma unroll
    for (int g = 0; g < 8; ++g) wgt[h][g] = e[g] * inv;
  }
  __syncthreads();

  unsigned short* op = O + (size_t)t * 1024;
#pragma unroll
  for (int h = 0; h < 8; ++h) {
    float a = 0.f;
#pragma unroll
    for (int g = 0; g < 8; ++g) a += wgt[h][g] * vv[g][tid];
    op[h * 128 + tid] = f2bf(a);
  }
}

// ---------------------------------------------------------------------------
// launch: Qproj -> KVproj -> attn -> outproj(+resid), twice.
// Workspace: Qbuf 64MB | KVbuf 128MB | Abuf 64MB  (256MB total)
// ---------------------------------------------------------------------------
extern "C" void kernel_launch(void* const* d_in, const int* in_sizes, int n_in,
                              void* d_out, int out_size, void* d_ws, size_t ws_size,
                              hipStream_t stream) {
  const float* liquid = (const float*)d_in[0];
  const float* mamba  = (const float*)d_in[1];
  const float* Wlq  = (const float*)d_in[2];
  const float* blq  = (const float*)d_in[3];
  const float* Wmkv = (const float*)d_in[4];
  const float* bmkv = (const float*)d_in[5];
  const float* Wmq  = (const float*)d_in[6];
  const float* bmq  = (const float*)d_in[7];
  const float* Wlkv = (const float*)d_in[8];
  const float* blkv = (const float*)d_in[9];
  const float* Wlo  = (const float*)d_in[10];
  const float* blo  = (const float*)d_in[11];
  const float* Wmo  = (const float*)d_in[12];
  const float* bmo  = (const float*)d_in[13];

  float* outL = (float*)d_out;
  float* outM = outL + (size_t)BT_TOKENS * 1024;

  char* ws = (char*)d_ws;
  unsigned short* Qbuf  = (unsigned short*)ws;                          // 64 MB
  unsigned short* KVbuf = (unsigned short*)(ws + ((size_t)64  << 20));  // 128 MB
  unsigned short* Abuf  = (unsigned short*)(ws + ((size_t)192 << 20));  // 64 MB

  const int M = BT_TOKENS, K = KDIM;
  dim3 blk(256);
  dim3 gN1(M / BM, 1024 / BN);
  dim3 gN2(M / BM, 2048 / BN);

  // ---- direction 1: liquid queries attend to mamba K/V ----
  gemm_proj<<<gN1, blk, 0, stream>>>(liquid, Wlq, blq, Qbuf,  M, 1024, K);
  gemm_proj<<<gN2, blk, 0, stream>>>(mamba, Wmkv, bmkv, KVbuf, M, 2048, K);
  head_attn<<<M, 128, 0, stream>>>(Qbuf, KVbuf, Abuf);
  gemm_out<<<gN1, blk, 0, stream>>>(Abuf, Wlo, blo, liquid, outL, M, 1024, K);

  // ---- direction 2: mamba queries attend to liquid K/V ----
  gemm_proj<<<gN1, blk, 0, stream>>>(mamba, Wmq, bmq, Qbuf,  M, 1024, K);
  gemm_proj<<<gN2, blk, 0, stream>>>(liquid, Wlkv, blkv, KVbuf, M, 2048, K);
  head_attn<<<M, 128, 0, stream>>>(Qbuf, KVbuf, Abuf);
  gemm_out<<<gN1, blk, 0, stream>>>(Abuf, Wmo, bmo, mamba, outM, M, 1024, K);
}
